// WanSelfAttention_62345745269159
// MI455X (gfx1250) — matrix operand order repaired and checked by reading, not compiled, not run on hardware
//
#include <hip/hip_runtime.h>

#define DIM   1536
#define S_LEN 4096
#define NH    12
#define HD_   128
#define NPAIR 64

typedef __attribute__((ext_vector_type(8)))  __bf16 v8bf;
typedef __attribute__((ext_vector_type(16))) __bf16 v16bf;
typedef __attribute__((ext_vector_type(8)))  float  v8f;

typedef unsigned short u16;
typedef unsigned int   u32;

__device__ __forceinline__ u16 f2bf(float f) {
  u32 u = __float_as_uint(f);
  u += 0x7FFFu + ((u >> 16) & 1u);   // round-to-nearest-even
  return (u16)(u >> 16);
}

__device__ __forceinline__ v16bf cat16(v8bf lo, v8bf hi) {
  return __builtin_shufflevector(lo, hi, 0, 1, 2, 3, 4, 5, 6, 7,
                                         8, 9, 10, 11, 12, 13, 14, 15);
}

__device__ __forceinline__ v8f vzero8() {
  v8f z = {0.f, 0.f, 0.f, 0.f, 0.f, 0.f, 0.f, 0.f};
  return z;
}

__device__ __forceinline__ v8f wmma_bf16(v16bf a, v16bf b, v8f c) {
  return __builtin_amdgcn_wmma_f32_16x16x32_bf16(false, a, false, b,
                                                 (short)0, c, false, false);
}

// scheduling fence: keep batched operand loads ahead of the MAC chain so the
// allocator must keep them live in distinct regs -> graduated s_wait_dscnt
__device__ __forceinline__ void sched_fence() {
#if __has_builtin(__builtin_amdgcn_sched_barrier)
  __builtin_amdgcn_sched_barrier(0);
#endif
}

// ---- CDNA5 async global->LDS copy (ASYNCcnt path) ----
// LDS byte address from the ldsoff VGPR; global address is SGPR base plus VGPR voff (GVS mode)
__device__ __forceinline__ void async_ld_b128(u32 ldsoff, const void* base, u32 voff) {
  asm volatile("global_load_async_to_lds_b128 %0, %1, %2"
               :: "v"(ldsoff), "v"(voff), "s"(base)
               : "memory");
}

template <int N>
__device__ __forceinline__ void async_wait() {
#if __has_builtin(__builtin_amdgcn_s_wait_asynccnt)
  __builtin_amdgcn_s_wait_asynccnt(N);
#else
  asm volatile("s_wait_asynccnt %0" :: "i"(N) : "memory");
#endif
}

__device__ __forceinline__ u32 lds_off_u32(const void* p) {
  // low 32 bits of a flat shared-aperture address == LDS byte offset
  return (u32)(uintptr_t)p;
}

// ---------------- elementwise casts / transposes ----------------

__global__ __launch_bounds__(256) void cast_bf16_kernel(const float* __restrict__ X,
                                                        u16* __restrict__ Y, int n) {
  int i = blockIdx.x * 256 + threadIdx.x;
  if (i < n) Y[i] = f2bf(X[i]);
}

// W [K,N] f32 row-major -> Wt [N,K] bf16 (so WMMA B loads are contiguous in K)
__global__ __launch_bounds__(256) void wtrans_kernel(const float* __restrict__ Wf,
                                                     u16* __restrict__ Wt) {
  int i = blockIdx.x * 256 + threadIdx.x;  // over DIM*DIM
  int k = i / DIM, n = i % DIM;
  Wt[(size_t)n * DIM + k] = f2bf(Wf[i]);
}

// V epilogue: Y[s, n*128+jd] f32 -> Vt[n][jd][s] bf16 (K-contiguous for PV GEMM)
__global__ __launch_bounds__(256) void vtrans_kernel(const float* __restrict__ Y,
                                                     u16* __restrict__ Vt) {
  int i = blockIdx.x * 256 + threadIdx.x;  // over S_LEN*DIM
  int s = i / DIM, j = i % DIM;
  int head = j >> 7, jd = j & 127;
  Vt[((size_t)head * HD_ + jd) * S_LEN + s] = f2bf(Y[i]);
}

// ---------------- WMMA GEMM with async-LDS B staging ----------------
// Y[M,N] f32 = A[M,K]bf16 @ B + bias, Bt is [N,K] bf16.
// block = 256 thr (8 waves); block tile 128x128; all waves share each 128x32 B
// tile, staged once per k-step into LDS (double buffered, ASYNCcnt prefetch).

__global__ __launch_bounds__(256) void gemm_bias_kernel(const u16* __restrict__ A,
                                                        const u16* __restrict__ Bt,
                                                        const float* __restrict__ bias,
                                                        float* __restrict__ Y) {
  __shared__ alignas(128) u16 btile[2][128 * 32];  // 2 x 8 KB

  const int tid  = threadIdx.x;
  const int wid  = tid >> 5;
  const int lane = tid & 31;
  const int half = lane >> 4;
  const int l16  = lane & 15;
  const int nt   = blockIdx.x % (DIM / 128);
  const int mt   = blockIdx.x / (DIM / 128);
  const int m0   = mt * 128 + wid * 16;
  const int n0   = nt * 128;

  // staging chunks: 128 rows x 64B = 512 x 16B; thread handles c and c+256
  const u32 c0 = (u32)tid;
  const u32 c1 = (u32)tid + 256u;
  const u32 gv0 = (c0 >> 2) * (DIM * 2) + (c0 & 3) * 16;  // row*3072 + sub*16
  const u32 gv1 = (c1 >> 2) * (DIM * 2) + (c1 & 3) * 16;
  const u32 lb  = lds_off_u32(&btile[0][0]);
  const char* bbase = (const char*)(Bt + (size_t)n0 * DIM);

  // prefetch k0 = 0 into buffer 0
  async_ld_b128(lb + c0 * 16, bbase, gv0);
  async_ld_b128(lb + c1 * 16, bbase, gv1);

  v8f acc[8];
#pragma unroll
  for (int j = 0; j < 8; ++j) acc[j] = vzero8();

  const u16* arow = A + (size_t)(m0 + l16) * DIM;
#pragma unroll 2
  for (int k0 = 0; k0 < DIM; k0 += 32) {
    const int buf  = (k0 >> 5) & 1;
    const int nbuf = buf ^ 1;

    // A operand from global (per-wave private rows)
    const int qa = k0 + half * 16;
    v16bf av = *(const v16bf*)(arow + qa);

    // prefetch next B tile; retire previous tile's loads (leave 2 outstanding)
    if (k0 + 32 < DIM) {
      const char* nb = bbase + (size_t)(k0 + 32) * 2;
      async_ld_b128(lb + (u32)nbuf * 8192u + c0 * 16, nb, gv0);
      async_ld_b128(lb + (u32)nbuf * 8192u + c1 * 16, nb, gv1);
      async_wait<2>();
    } else {
      async_wait<0>();
    }
    __syncthreads();

    // batch all B operand loads, then issue WMMAs -> graduated DS waits
    const u16* bt = &btile[buf][0];
    v16bf bv[8];
#pragma unroll
    for (int j = 0; j < 8; ++j)
      bv[j] = *(const v16bf*)(bt + (j * 16 + l16) * 32 + half * 16);
    sched_fence();  // all 8 LDS loads issued before any WMMA
#pragma unroll
    for (int j = 0; j < 8; ++j)
      acc[j] = wmma_bf16(av, bv[j], acc[j]);
    __syncthreads();  // reads done before next prefetch overwrites this buffer
  }

  // C layout: col = l16, row = r + half*8
#pragma unroll
  for (int j = 0; j < 8; ++j) {
    const int col = n0 + j * 16 + l16;
    const float b = bias[col];
#pragma unroll
    for (int r = 0; r < 8; ++r)
      Y[(size_t)(m0 + r + half * 8) * DIM + col] = acc[j][r] + b;
  }
}

// ---------------- RMSNorm + 3D RoPE, f32 -> bf16 [head][S][128] ----------------

__global__ __launch_bounds__(256) void rmsrope_kernel(const float* __restrict__ Y,
                                                      const float* __restrict__ g,
                                                      const float* __restrict__ fcos,
                                                      const float* __restrict__ fsin,
                                                      const int* __restrict__ Fp,
                                                      const int* __restrict__ Hp,
                                                      const int* __restrict__ Wp,
                                                      u16* __restrict__ Obf) {
  const int s = blockIdx.x;
  const int t = threadIdx.x;
  const float* row = Y + (size_t)s * DIM;
  float vals[6];
  float ss = 0.f;
#pragma unroll
  for (int i = 0; i < 6; ++i) {
    float v = row[t * 6 + i];
    vals[i] = v;
    ss += v * v;
  }
#pragma unroll
  for (int off = 16; off >= 1; off >>= 1) ss += __shfl_xor(ss, off, 32);
  __shared__ float red[8];
  if ((t & 31) == 0) red[t >> 5] = ss;
  __syncthreads();
  float tot = 0.f;
#pragma unroll
  for (int i = 0; i < 8; ++i) tot += red[i];
  const float inv = rsqrtf(tot * (1.0f / DIM) + 1e-6f);

  const int Wd = Wp[0], Hd = Hp[0];
  (void)Fp;
  const int w = s % Wd;
  const int h = (s / Wd) % Hd;
  const int f = s / (Wd * Hd);
  // c=64: pairs [0,22) temporal, [22,43) height, [43,64) width (absolute col)
#pragma unroll
  for (int p = 0; p < 3; ++p) {
    const int j    = t * 6 + p * 2;
    const int head = j >> 7;
    const int jd   = j & 127;
    const int pr   = jd >> 1;
    const int pos  = (pr < 22) ? f : ((pr < 43) ? h : w);
    const float cs = fcos[pos * NPAIR + pr];
    const float sn = fsin[pos * NPAIR + pr];
    const float xr = vals[p * 2]     * inv * g[j];
    const float xi = vals[p * 2 + 1] * inv * g[j + 1];
    u16* dst = Obf + ((size_t)head * S_LEN + s) * HD_ + jd;
    dst[0] = f2bf(xr * cs - xi * sn);
    dst[1] = f2bf(xr * sn + xi * cs);
  }
}

// ---------------- Flash attention with async-LDS K/V staging ----------------
// Q,K: [12][S][128] bf16 ; Vt: [12][128][S] bf16 ; O: [S][1536] bf16
// block = 8 waves x 16 q rows = 128 q rows per head; K/V tiles shared by all
// waves, staged via ASYNCcnt double buffering.

__global__ __launch_bounds__(256) void attn_kernel(const u16* __restrict__ Q,
                                                   const u16* __restrict__ K,
                                                   const u16* __restrict__ Vt,
                                                   const int* __restrict__ seq_lens,
                                                   u16* __restrict__ O) {
  __shared__ alignas(128) u16 ktile[2][32 * 128];  // 2 x 8 KB, [key][d]
  __shared__ alignas(128) u16 vtile[2][128 * 32];  // 2 x 8 KB, [d][key]
  __shared__ alignas(128) u16 pbuf[8][16 * 32];    // per-wave P re-layout

  const int tid  = threadIdx.x;
  const int wid  = tid >> 5;
  const int lane = tid & 31;
  const int half = lane >> 4;
  const int l16  = lane & 15;
  const int head = blockIdx.x / (S_LEN / 128);
  const int qt   = blockIdx.x % (S_LEN / 128);
  const int q0   = qt * 128 + wid * 16;
  const int slen = seq_lens[0];
  const float scale = 0.08838834764831845f;  // 1/sqrt(128)

  const u16* Qh = Q  + (size_t)head * S_LEN * HD_;
  const u16* Kh = K  + (size_t)head * S_LEN * HD_;
  const u16* Vh = Vt + (size_t)head * HD_ * S_LEN;

  // staging chunks (512 x 16B per tile); thread handles c0 and c1
  const u32 c0  = (u32)tid;
  const u32 c1  = (u32)tid + 256u;
  const u32 vv0 = (c0 >> 2) * (S_LEN * 2) + (c0 & 3) * 16;  // Vt row stride 8192B
  const u32 vv1 = (c1 >> 2) * (S_LEN * 2) + (c1 & 3) * 16;
  const u32 klb = lds_off_u32(&ktile[0][0]);
  const u32 vlb = lds_off_u32(&vtile[0][0]);

  // prefetch kt = 0 into buffer 0 (K tile is one contiguous 8 KB block)
  {
    const char* kb = (const char*)Kh;
    const char* vb = (const char*)Vh;
    async_ld_b128(klb + c0 * 16, kb, c0 * 16);
    async_ld_b128(klb + c1 * 16, kb, c1 * 16);
    async_ld_b128(vlb + c0 * 16, vb, vv0);
    async_ld_b128(vlb + c1 * 16, vb, vv1);
  }

  // Q A-tiles for d = 4 x 32, held in registers for the whole key loop
  v16bf qa[4];
  const u16* qrow = Qh + (size_t)(q0 + l16) * HD_;
#pragma unroll
  for (int ks = 0; ks < 4; ++ks) {
    const int qo = ks * 32 + half * 16;
    qa[ks] = *(const v16bf*)(qrow + qo);
  }

  v8f o[8];
#pragma unroll
  for (int j = 0; j < 8; ++j) o[j] = vzero8();
  float m[8], l[8];
#pragma unroll
  for (int r = 0; r < 8; ++r) { m[r] = -1e30f; l[r] = 0.f; }

  u16* myp = &pbuf[wid][0];

#pragma unroll 1
  for (int kt = 0; kt < S_LEN; kt += 32) {
    const int buf  = (kt >> 5) & 1;
    const int nbuf = buf ^ 1;

    // prefetch next K/V tiles; retire previous (leave 4 outstanding)
    if (kt + 32 < S_LEN) {
      const char* kb = (const char*)Kh + (size_t)(kt + 32) * 256;
      const char* vb = (const char*)Vh + (size_t)(kt + 32) * 2;
      async_ld_b128(klb + (u32)nbuf * 8192u + c0 * 16, kb, c0 * 16);
      async_ld_b128(klb + (u32)nbuf * 8192u + c1 * 16, kb, c1 * 16);
      async_ld_b128(vlb + (u32)nbuf * 8192u + c0 * 16, vb, vv0);
      async_ld_b128(vlb + (u32)nbuf * 8192u + c1 * 16, vb, vv1);
      async_wait<4>();
    } else {
      async_wait<0>();
    }
    __syncthreads();

    // ---- scores: 16q x 32k as two 16x16 tiles, K operand from LDS ----
    // batch all LDS loads, then WMMAs -> graduated DS waits
    const u16* kt_lds = &ktile[buf][0];
    v16bf kb0[4], kb1[4];
#pragma unroll
    for (int ks = 0; ks < 4; ++ks) {
      const int kb = ks * 32 + half * 16;
      kb0[ks] = *(const v16bf*)(kt_lds + l16 * 128 + kb);
      kb1[ks] = *(const v16bf*)(kt_lds + (16 + l16) * 128 + kb);
    }
    sched_fence();  // all 8 LDS loads issued before any WMMA
    v8f s0 = vzero8(), s1 = vzero8();
#pragma unroll
    for (int ks = 0; ks < 4; ++ks) {
      s0 = wmma_bf16(qa[ks], kb0[ks], s0);
      s1 = wmma_bf16(qa[ks], kb1[ks], s1);
    }
    const bool ok0 = (kt + l16) < slen;
    const bool ok1 = (kt + 16 + l16) < slen;

    // ---- online softmax: row r+half*8 lives entirely in this 16-lane half ----
#pragma unroll
    for (int r = 0; r < 8; ++r) {
      float a = ok0 ? s0[r] * scale : -1e30f;
      float b = ok1 ? s1[r] * scale : -1e30f;
      float mx = fmaxf(a, b);
#pragma unroll
      for (int off = 8; off >= 1; off >>= 1) mx = fmaxf(mx, __shfl_xor(mx, off, 32));
      const float mn = fmaxf(m[r], mx);
      const float sc = __expf(m[r] - mn);
      const float p0 = __expf(a - mn);
      const float p1 = __expf(b - mn);
      float rs = p0 + p1;
#pragma unroll
      for (int off = 8; off >= 1; off >>= 1) rs += __shfl_xor(rs, off, 32);
      l[r] = l[r] * sc + rs;
      m[r] = mn;
#pragma unroll
      for (int j = 0; j < 8; ++j) o[j][r] *= sc;  // rescale O (same row mapping)
      myp[(r + half * 8) * 32 + l16]      = f2bf(p0);
      myp[(r + half * 8) * 32 + 16 + l16] = f2bf(p1);
    }

    // ---- read P back in A-layout (same-wave DS ops are in-order) ----
    const u16* prow = myp + l16 * 32;
    const int qo = half * 16;
    v16bf pa = *(const v16bf*)(prow + qo);

    // ---- O += P(16x32) @ V(32x128), V operand from LDS ----
    const u16* vt_lds = &vtile[buf][0];
    v16bf vvv[8];
#pragma unroll
    for (int j = 0; j < 8; ++j)
      vvv[j] = *(const v16bf*)(vt_lds + (j * 16 + l16) * 32 + half * 16);
    sched_fence();  // all 8 LDS loads issued before any WMMA
#pragma unroll
    for (int j = 0; j < 8; ++j)
      o[j] = wmma_bf16(pa, vvv[j], o[j]);
    __syncthreads();  // all waves done with buf before it is overwritten
  }

  // ---- normalize and store bf16 [S][1536] for the output projection ----
#pragma unroll
  for (int r = 0; r < 8; ++r) {
    const float invl = 1.0f / l[r];
    const int row = q0 + r + half * 8;
#pragma unroll
    for (int j = 0; j < 8; ++j)
      O[(size_t)row * DIM + head * HD_ + j * 16 + l16] = f2bf(o[j][r] * invl);
  }
}

// ---------------- host launcher ----------------

extern "C" void kernel_launch(void* const* d_in, const int* in_sizes, int n_in,
                              void* d_out, int out_size, void* d_ws, size_t ws_size,
                              hipStream_t stream) {
  (void)in_sizes; (void)n_in; (void)out_size; (void)ws_size;
  const float* x  = (const float*)d_in[0];
  const int*   sl = (const int*)d_in[1];
  const int*   Fp = (const int*)d_in[2];
  const int*   Hp = (const int*)d_in[3];
  const int*   Wp = (const int*)d_in[4];
  const float* Wq = (const float*)d_in[5];
  const float* bq = (const float*)d_in[6];
  const float* Wk = (const float*)d_in[7];
  const float* bk = (const float*)d_in[8];
  const float* Wv = (const float*)d_in[9];
  const float* bv = (const float*)d_in[10];
  const float* Wo = (const float*)d_in[11];
  const float* bo = (const float*)d_in[12];
  const float* gq = (const float*)d_in[13];
  const float* gk = (const float*)d_in[14];
  const float* fc = (const float*)d_in[15];
  const float* fs = (const float*)d_in[16];

  char* ws = (char*)d_ws;
  size_t off = 0;
  auto take = [&](size_t bytes) -> char* {
    char* p = ws + off;
    off += (bytes + 255) & ~(size_t)255;
    return p;
  };
  u16*   xbf = (u16*)take((size_t)S_LEN * DIM * 2);
  u16*   wtq = (u16*)take((size_t)DIM * DIM * 2);
  u16*   wtk = (u16*)take((size_t)DIM * DIM * 2);
  u16*   wtv = (u16*)take((size_t)DIM * DIM * 2);
  u16*   wto = (u16*)take((size_t)DIM * DIM * 2);
  float* Yf  = (float*)take((size_t)S_LEN * DIM * 4);  // reused for q,k,v raw GEMM out
  u16*   qbf = (u16*)take((size_t)NH * S_LEN * HD_ * 2);
  u16*   kbf = (u16*)take((size_t)NH * S_LEN * HD_ * 2);
  u16*   vtb = (u16*)take((size_t)NH * HD_ * S_LEN * 2);
  u16*   abf = (u16*)take((size_t)S_LEN * DIM * 2);

  const int nelem = S_LEN * DIM;
  const int welem = DIM * DIM;
  cast_bf16_kernel<<<nelem / 256, 256, 0, stream>>>(x, xbf, nelem);
  wtrans_kernel<<<welem / 256, 256, 0, stream>>>(Wq, wtq);
  wtrans_kernel<<<welem / 256, 256, 0, stream>>>(Wk, wtk);
  wtrans_kernel<<<welem / 256, 256, 0, stream>>>(Wv, wtv);
  wtrans_kernel<<<welem / 256, 256, 0, stream>>>(Wo, wto);

  dim3 gg((S_LEN / 128) * (DIM / 128));
  gemm_bias_kernel<<<gg, 256, 0, stream>>>(xbf, wtq, bq, Yf);
  rmsrope_kernel<<<S_LEN, 256, 0, stream>>>(Yf, gq, fc, fs, Fp, Hp, Wp, qbf);
  gemm_bias_kernel<<<gg, 256, 0, stream>>>(xbf, wtk, bk, Yf);
  rmsrope_kernel<<<S_LEN, 256, 0, stream>>>(Yf, gk, fc, fs, Fp, Hp, Wp, kbf);
  gemm_bias_kernel<<<gg, 256, 0, stream>>>(xbf, wtv, bv, Yf);
  vtrans_kernel<<<nelem / 256, 256, 0, stream>>>(Yf, vtb);

  attn_kernel<<<NH * (S_LEN / 128), 256, 0, stream>>>(qbf, kbf, vtb, sl, abf);

  gemm_bias_kernel<<<gg, 256, 0, stream>>>(abf, wto, bo, (float*)d_out);
}
